// CausalSelfAttention_37056977830442
// MI455X (gfx1250) — compile-verified, matrix operand order checked by reading, not run on hardware
//
#include <hip/hip_runtime.h>
#include <hip/hip_bf16.h>
#include <stdint.h>

// ---------------------------------------------------------------------------
// MI455X / gfx1250 causal self-attention (GQA + RoPE), bf16 WMMA pipeline.
//   fp32 inputs -> bf16 (weights transposed) -> WMMA GEMMs (f32 accum)
//   -> RoPE (+score scale folded into Q) -> flash attention (WMMA QK^T and PV)
//   -> WMMA output projection -> fp32 out.
// CDNA5 async global->LDS copies (GLOBAL_LOAD_ASYNC_TO_LDS_B128, ASYNCcnt)
// with double-buffered GEMM K-loop; sync fallback kept under #if.
// Workspace use: ~289 MB.
// ---------------------------------------------------------------------------

typedef __attribute__((ext_vector_type(16))) __bf16        bf16x16;
typedef __attribute__((ext_vector_type(8)))  float         floatx8;
typedef __attribute__((ext_vector_type(4)))  unsigned int  uint32x4;

#define DIM    4096
#define NH     32
#define NKVH   8
#define HD     128
#define BSZ    2
#define SEQ    2048
#define MROWS  (BSZ * SEQ)      // 4096
#define KVDIM  (NKVH * HD)      // 1024

#if defined(__HIP_DEVICE_COMPILE__) && defined(__has_builtin)
#if __has_builtin(__builtin_amdgcn_global_load_async_to_lds_b128)
#define HAVE_ASYNC_LDS 1
#endif
#endif

#ifdef HAVE_ASYNC_LDS
// Per-lane async DMA of 16B from global to LDS; tracked by ASYNCcnt, no VGPR
// round-trip (CDNA5 ISA).  Builtin parameter type per
// clang diagnostic: `int __vector(4) *` in AS(1)/AS(3), imm offset, imm cpol.
typedef int v4i_gcc __attribute__((vector_size(16)));
typedef __attribute__((address_space(1))) v4i_gcc* g_v4i_ptr;
typedef __attribute__((address_space(3))) v4i_gcc* l_v4i_ptr;

__device__ __forceinline__ void async_g2l_b128(const void* g, void* l) {
    void* gnc = const_cast<void*>(g);
    __attribute__((address_space(1))) void* g1 =
        (__attribute__((address_space(1))) void*)gnc;
    __attribute__((address_space(3))) void* l3 =
        (__attribute__((address_space(3))) void*)l;
    __builtin_amdgcn_global_load_async_to_lds_b128((g_v4i_ptr)g1, (l_v4i_ptr)l3,
                                                   /*offset=*/0, /*cpol=*/0);
}
__device__ __forceinline__ void wait_async0() {
    asm volatile("s_wait_asynccnt 0" ::: "memory");
}
#endif

__device__ __forceinline__ unsigned short f2bf(float f) {
    unsigned int u = __float_as_uint(f);
    u += 0x7FFFu + ((u >> 16) & 1u);       // round-to-nearest-even
    return (unsigned short)(u >> 16);
}

__device__ __forceinline__ floatx8 fzero8() {
    floatx8 z;
#pragma unroll
    for (int i = 0; i < 8; ++i) z[i] = 0.0f;
    return z;
}

// Load a 16x32 bf16 fragment in the CDNA5 WMMA A-operand layout from an LDS
// tile (row-major, `stride` elements per row).  Per ISA 7.12.2:
//   lane l holds row m = l%16; K values { koff..koff+7 } U { 16+koff..16+koff+7 }
//   with koff = 8*(l/16).
__device__ __forceinline__ bf16x16 ldsA(const unsigned short* t, int row0,
                                        int col0, int stride, int lane) {
    int r  = row0 + (lane & 15);
    int kb = col0 + ((lane >> 4) << 3);
    union { uint32x4 u[2]; bf16x16 v; } f;
    f.u[0] = *(const uint32x4*)(t + (size_t)r * stride + kb);
    f.u[1] = *(const uint32x4*)(t + (size_t)r * stride + kb + 16);
    return f.v;
}

__device__ __forceinline__ floatx8 wmma_bf16(bf16x16 a, bf16x16 b, floatx8 c) {
    // D = A(16x32) * B(32x16) + C, f32 accumulate
    return __builtin_amdgcn_wmma_f32_16x16x32_bf16(false, a, false, b,
                                                   (short)0, c, false, false);
}

// ---------------------------------------------------------------------------
// Elementwise fp32 -> bf16
// ---------------------------------------------------------------------------
__global__ void cvt_bf16_kernel(const float* __restrict__ s,
                                unsigned short* __restrict__ d, size_t n) {
    size_t i = (size_t)blockIdx.x * blockDim.x + threadIdx.x;
    if (i < n) d[i] = f2bf(s[i]);
}

// Transpose-convert: src is (K x N) row-major fp32, dst is (N x K) bf16.
__global__ void cvtT_bf16_kernel(const float* __restrict__ s,
                                 unsigned short* __restrict__ d, int K, int N) {
    size_t i = (size_t)blockIdx.x * blockDim.x + threadIdx.x;
    if (i >= (size_t)K * N) return;
    size_t nn = i / (size_t)K, kk = i % (size_t)K;
    d[i] = f2bf(s[kk * (size_t)N + nn]);
}

// ---------------------------------------------------------------------------
// GEMM: C(MxN,f32) = A(MxK,bf16 row-major) * BT(NxK,bf16 row-major)^T
// 128x128 block tile, BK=64, 256 threads = 8 waves in a 4(M) x 2(N) grid,
// each wave owns 32x64 of C (2x4 WMMA accumulators).
// Async path: ping-pong LDS buffers; the (k+1) tile streams in via the async
// DMA while the current tile's 16 WMMAs execute; one barrier per K-step.
// ---------------------------------------------------------------------------
#define BM 128
#define BN 128
#define BK 64

__global__ __launch_bounds__(256) void gemm_bf16_wmma(
    const unsigned short* __restrict__ A,
    const unsigned short* __restrict__ BT,
    float* __restrict__ C, int M, int N, int K) {
    const int tid = threadIdx.x, lane = tid & 31, w = tid >> 5;
    const int wm = w & 3, wn = w >> 2;
    const int m0 = blockIdx.y * BM, n0 = blockIdx.x * BN;

    floatx8 acc[2][4];
#pragma unroll
    for (int mi = 0; mi < 2; ++mi)
#pragma unroll
        for (int ni = 0; ni < 4; ++ni) acc[mi][ni] = fzero8();

#ifdef HAVE_ASYNC_LDS
    __shared__ unsigned short As[2][BM * BK];
    __shared__ unsigned short Bs[2][BN * BK];

    auto issue = [&](int k0, int buf) {
#pragma unroll
        for (int i = 0; i < 4; ++i) {
            int idx = tid + i * 256;
            int r = idx >> 3, c8 = (idx & 7) << 3;
            async_g2l_b128(A + (size_t)(m0 + r) * K + k0 + c8,
                           &As[buf][r * BK + c8]);
            async_g2l_b128(BT + (size_t)(n0 + r) * K + k0 + c8,
                           &Bs[buf][r * BK + c8]);
        }
    };

    issue(0, 0);
    wait_async0();
    __syncthreads();
    const int nk = K / BK;
    for (int kt = 0; kt < nk; ++kt) {
        const int buf = kt & 1;
        if (kt + 1 < nk) issue((kt + 1) * BK, buf ^ 1);  // overlap with WMMA
        const unsigned short* Ab = &As[buf][0];
        const unsigned short* Bb = &Bs[buf][0];
#pragma unroll
        for (int kc = 0; kc < BK / 32; ++kc) {
            bf16x16 af[2], bfm[4];
            af[0] = ldsA(Ab, wm * 32, kc * 32, BK, lane);
            af[1] = ldsA(Ab, wm * 32 + 16, kc * 32, BK, lane);
#pragma unroll
            for (int ni = 0; ni < 4; ++ni)
                bfm[ni] = ldsA(Bb, wn * 64 + ni * 16, kc * 32, BK, lane);
#pragma unroll
            for (int mi = 0; mi < 2; ++mi)
#pragma unroll
                for (int ni = 0; ni < 4; ++ni)
                    acc[mi][ni] = wmma_bf16(af[mi], bfm[ni], acc[mi][ni]);
        }
        wait_async0();     // (k+1) tile landed
        __syncthreads();   // everyone done reading buf, tile visible
    }
#else
    __shared__ unsigned short As[BM * BK];
    __shared__ unsigned short Bs[BN * BK];
    for (int k0 = 0; k0 < K; k0 += BK) {
        __syncthreads();
#pragma unroll
        for (int i = 0; i < 4; ++i) {
            int idx = tid + i * 256;
            int r = idx >> 3, c8 = (idx & 7) << 3;
            *(uint32x4*)(As + r * BK + c8) =
                *(const uint32x4*)(A + (size_t)(m0 + r) * K + k0 + c8);
            *(uint32x4*)(Bs + r * BK + c8) =
                *(const uint32x4*)(BT + (size_t)(n0 + r) * K + k0 + c8);
        }
        if (k0 + BK < K) {
            int r = tid >> 3, c8 = (tid & 7) << 3;
            __builtin_prefetch(A + (size_t)(m0 + r) * K + k0 + BK + c8, 0, 1);
            __builtin_prefetch(BT + (size_t)(n0 + r) * K + k0 + BK + c8, 0, 1);
        }
        __syncthreads();
#pragma unroll
        for (int kc = 0; kc < BK / 32; ++kc) {
            bf16x16 af[2], bfm[4];
            af[0] = ldsA(As, wm * 32, kc * 32, BK, lane);
            af[1] = ldsA(As, wm * 32 + 16, kc * 32, BK, lane);
#pragma unroll
            for (int ni = 0; ni < 4; ++ni)
                bfm[ni] = ldsA(Bs, wn * 64 + ni * 16, kc * 32, BK, lane);
#pragma unroll
            for (int mi = 0; mi < 2; ++mi)
#pragma unroll
                for (int ni = 0; ni < 4; ++ni)
                    acc[mi][ni] = wmma_bf16(af[mi], bfm[ni], acc[mi][ni]);
        }
    }
#endif

    // C/D layout: VGPR g, lane l -> row = g + 8*(l>>4), col = l&15 (ISA 7.12.2)
    const int rg = (lane >> 4) << 3, cl = lane & 15;
#pragma unroll
    for (int mi = 0; mi < 2; ++mi)
#pragma unroll
        for (int ni = 0; ni < 4; ++ni)
#pragma unroll
            for (int g = 0; g < 8; ++g)
                C[(size_t)(m0 + wm * 32 + mi * 16 + rg + g) * N +
                  (n0 + wn * 64 + ni * 16 + cl)] = acc[mi][ni][g];
}

// ---------------------------------------------------------------------------
// RoPE + bf16 convert.  src/dst layout (B*S, nheads*HD); scale folds the
// 1/sqrt(HD) attention scale into Q.  One thread per (row, head, pair).
// ---------------------------------------------------------------------------
__global__ void rope_cvt_kernel(const float* __restrict__ s,
                                unsigned short* __restrict__ d,
                                const float* __restrict__ fc,
                                const float* __restrict__ fs,
                                int nheads, float scale, size_t n) {
    size_t i = (size_t)blockIdx.x * blockDim.x + threadIdx.x;
    if (i >= n) return;
    int p = (int)(i & 63);
    int h = (int)((i >> 6) % (size_t)nheads);
    size_t row = i / ((size_t)nheads << 6);
    int sq = (int)(row & (SEQ - 1));
    size_t base = row * ((size_t)nheads * HD) + (size_t)h * HD + 2 * p;
    float tr = s[base], ti = s[base + 1];
    float c = fc[sq * 64 + p], sn = fs[sq * 64 + p];
    d[base]     = f2bf((tr * c - ti * sn) * scale);
    d[base + 1] = f2bf((tr * sn + ti * c) * scale);
}

// V: (B*S, NKVH*HD) f32 -> (B, NKVH, HD, S) bf16 (d-major so PV's B operand
// loads with the A-layout loader).
__global__ void vtrans_kernel(const float* __restrict__ v,
                              unsigned short* __restrict__ vt, size_t n) {
    size_t i = (size_t)blockIdx.x * blockDim.x + threadIdx.x;
    if (i >= n) return;
    int s  = (int)(i & (SEQ - 1));
    int dd = (int)((i >> 11) & (HD - 1));
    int h  = (int)((i >> 18) & (NKVH - 1));
    int b  = (int)(i >> 21);
    vt[i] = f2bf(v[((size_t)(b * SEQ + s)) * KVDIM + h * HD + dd]);
}

// ---------------------------------------------------------------------------
// Flash attention, causal, GQA (4 q-heads per kv-head).
// Block = (64 q rows, one head, one batch), 4 waves x 16 rows.
// Q pre-scaled by 1/sqrt(HD).  Online softmax in fp32, P re-staged as bf16.
// K/V tiles stream in via the async global->LDS DMA when available.
// ---------------------------------------------------------------------------
#define BQ  64
#define BKV 64

__global__ __launch_bounds__(128) void attn_fwd_kernel(
    const unsigned short* __restrict__ q,   // (B*S, NH*HD)   bf16, pre-scaled
    const unsigned short* __restrict__ k,   // (B*S, NKVH*HD) bf16
    const unsigned short* __restrict__ vt,  // (B, NKVH, HD, S) bf16
    unsigned short* __restrict__ o) {       // (B*S, NH*HD)   bf16
    __shared__ unsigned short Qs[BQ * HD];      // 16 KB
    __shared__ unsigned short Ks[BKV * HD];     // 16 KB (rows = keys)
    __shared__ unsigned short Vs[HD * BKV];     // 16 KB (rows = d, V^T)
    __shared__ unsigned short Ps[4][16 * BKV];  //  8 KB (wave-private P)

    const int tid = threadIdx.x, lane = tid & 31, w = tid >> 5;
    const int qt = blockIdx.x, h = blockIdx.y, b = blockIdx.z;
    const int kvh = h >> 2;                 // N_REP = 4
    const int q0 = qt * BQ;

    // Stage Q tile, then pull this wave's 16x128 into 4 A-fragments.
#ifdef HAVE_ASYNC_LDS
#pragma unroll
    for (int i = 0; i < 8; ++i) {
        int idx = tid + i * 128;
        int r = idx >> 4, c8 = (idx & 15) << 3;
        async_g2l_b128(q + (size_t)(b * SEQ + q0 + r) * (NH * HD) + h * HD + c8,
                       Qs + r * HD + c8);
    }
    wait_async0();
#else
#pragma unroll
    for (int i = 0; i < 8; ++i) {
        int idx = tid + i * 128;
        int r = idx >> 4, c8 = (idx & 15) << 3;
        *(uint32x4*)(Qs + r * HD + c8) =
            *(const uint32x4*)(q + (size_t)(b * SEQ + q0 + r) * (NH * HD) +
                               h * HD + c8);
    }
#endif
    __syncthreads();
    bf16x16 qf[4];
#pragma unroll
    for (int kc = 0; kc < 4; ++kc) qf[kc] = ldsA(Qs, w * 16, kc * 32, HD, lane);

    floatx8 O[8];
#pragma unroll
    for (int dt = 0; dt < 8; ++dt) O[dt] = fzero8();
    float mrow[8], lrow[8];
#pragma unroll
    for (int g = 0; g < 8; ++g) { mrow[g] = -__builtin_inff(); lrow[g] = 0.0f; }

    const int rbase = q0 + w * 16 + ((lane >> 4) << 3);   // C-layout row base

    for (int kv = 0; kv <= qt; ++kv) {
        const int kv0 = kv * BKV;
        __syncthreads();
#ifdef HAVE_ASYNC_LDS
#pragma unroll
        for (int i = 0; i < 8; ++i) {          // K tile: 64 keys x 128 d
            int idx = tid + i * 128;
            int r = idx >> 4, c8 = (idx & 15) << 3;
            async_g2l_b128(
                k + (size_t)(b * SEQ + kv0 + r) * KVDIM + kvh * HD + c8,
                Ks + r * HD + c8);
        }
#pragma unroll
        for (int i = 0; i < 8; ++i) {          // V^T tile: 128 d x 64 keys
            int idx = tid + i * 128;
            int dd = idx >> 3, s8 = (idx & 7) << 3;
            async_g2l_b128(
                vt + (((size_t)(b * NKVH + kvh)) * HD + dd) * SEQ + kv0 + s8,
                Vs + dd * BKV + s8);
        }
        wait_async0();
#else
#pragma unroll
        for (int i = 0; i < 8; ++i) {
            int idx = tid + i * 128;
            int r = idx >> 4, c8 = (idx & 15) << 3;
            *(uint32x4*)(Ks + r * HD + c8) =
                *(const uint32x4*)(k + (size_t)(b * SEQ + kv0 + r) * KVDIM +
                                   kvh * HD + c8);
        }
#pragma unroll
        for (int i = 0; i < 8; ++i) {
            int idx = tid + i * 128;
            int dd = idx >> 3, s8 = (idx & 7) << 3;
            *(uint32x4*)(Vs + dd * BKV + s8) =
                *(const uint32x4*)(vt + (((size_t)(b * NKVH + kvh)) * HD + dd) *
                                            SEQ + kv0 + s8);
        }
#endif
        __syncthreads();

        // scores = Q * K^T  (B-layout of K^T == A-layout of K tile)
        floatx8 sc[4];
#pragma unroll
        for (int nt = 0; nt < 4; ++nt) {
            sc[nt] = fzero8();
#pragma unroll
            for (int kc = 0; kc < 4; ++kc)
                sc[nt] = wmma_bf16(qf[kc],
                                   ldsA(Ks, nt * 16, kc * 32, HD, lane), sc[nt]);
        }

        if (kv == qt) {  // diagonal tile -> causal mask (branch-free per lane)
#pragma unroll
            for (int nt = 0; nt < 4; ++nt) {
                int col = kv0 + nt * 16 + (lane & 15);
#pragma unroll
                for (int g = 0; g < 8; ++g)
                    sc[nt][g] = (col > rbase + g) ? -__builtin_inff() : sc[nt][g];
            }
        }

        // Online softmax: row reductions across the 16-lane half-groups.
        float fac[8];
#pragma unroll
        for (int g = 0; g < 8; ++g) {
            float t = fmaxf(fmaxf(sc[0][g], sc[1][g]), fmaxf(sc[2][g], sc[3][g]));
            t = fmaxf(t, __shfl_xor(t, 1, 32));
            t = fmaxf(t, __shfl_xor(t, 2, 32));
            t = fmaxf(t, __shfl_xor(t, 4, 32));
            t = fmaxf(t, __shfl_xor(t, 8, 32));
            float nm = fmaxf(mrow[g], t);
            fac[g] = __expf(mrow[g] - nm);
            mrow[g] = nm;
#pragma unroll
            for (int nt = 0; nt < 4; ++nt) sc[nt][g] = __expf(sc[nt][g] - nm);
            float rs = sc[0][g] + sc[1][g] + sc[2][g] + sc[3][g];
            rs += __shfl_xor(rs, 1, 32);
            rs += __shfl_xor(rs, 2, 32);
            rs += __shfl_xor(rs, 4, 32);
            rs += __shfl_xor(rs, 8, 32);
            lrow[g] = lrow[g] * fac[g] + rs;
        }
#pragma unroll
        for (int dt = 0; dt < 8; ++dt)
#pragma unroll
            for (int g = 0; g < 8; ++g) O[dt][g] *= fac[g];

        // Stage P (16x64 bf16, wave-private) and do O += P * V.
        unsigned short* pw = &Ps[w][0];
#pragma unroll
        for (int nt = 0; nt < 4; ++nt)
#pragma unroll
            for (int g = 0; g < 8; ++g)
                pw[(((lane >> 4) << 3) + g) * BKV + nt * 16 + (lane & 15)] =
                    f2bf(sc[nt][g]);
        __syncthreads();

        bf16x16 pf0 = ldsA(pw, 0, 0, BKV, lane);
        bf16x16 pf1 = ldsA(pw, 0, 32, BKV, lane);
#pragma unroll
        for (int dt = 0; dt < 8; ++dt) {
            O[dt] = wmma_bf16(pf0, ldsA(Vs, dt * 16, 0, BKV, lane), O[dt]);
            O[dt] = wmma_bf16(pf1, ldsA(Vs, dt * 16, 32, BKV, lane), O[dt]);
        }
    }

    // Normalize and write (b, s, h, d) bf16.
#pragma unroll
    for (int g = 0; g < 8; ++g) {
        float inv = 1.0f / lrow[g];
        int row = rbase + g;
#pragma unroll
        for (int dt = 0; dt < 8; ++dt)
            o[(size_t)(b * SEQ + row) * (NH * HD) + h * HD + dt * 16 +
              (lane & 15)] = f2bf(O[dt][g] * inv);
    }
}

// ---------------------------------------------------------------------------
// Host-side orchestration (all on `stream`, workspace-only scratch).
// ---------------------------------------------------------------------------
extern "C" void kernel_launch(void* const* d_in, const int* in_sizes, int n_in,
                              void* d_out, int out_size, void* d_ws,
                              size_t ws_size, hipStream_t stream) {
    (void)in_sizes; (void)n_in; (void)out_size; (void)ws_size;
    const float* x  = (const float*)d_in[0];
    const float* wq = (const float*)d_in[1];
    const float* wk = (const float*)d_in[2];
    const float* wv = (const float*)d_in[3];
    const float* wo = (const float*)d_in[4];
    const float* fc = (const float*)d_in[5];
    const float* fs = (const float*)d_in[6];

    char* wsp = (char*)d_ws;
    size_t off = 0;
    auto alloc = [&](size_t bytes) -> void* {
        void* p = wsp + off;
        off += (bytes + 255) & ~(size_t)255;
        return p;
    };
    unsigned short* xb  = (unsigned short*)alloc((size_t)MROWS * DIM * 2);
    unsigned short* wqT = (unsigned short*)alloc((size_t)DIM * DIM * 2);
    unsigned short* wkT = (unsigned short*)alloc((size_t)KVDIM * DIM * 2);
    unsigned short* wvT = (unsigned short*)alloc((size_t)KVDIM * DIM * 2);
    unsigned short* woT = (unsigned short*)alloc((size_t)DIM * DIM * 2);
    float*          qf  = (float*)alloc((size_t)MROWS * DIM * 4);
    float*          kf  = (float*)alloc((size_t)MROWS * KVDIM * 4);
    float*          vf  = (float*)alloc((size_t)MROWS * KVDIM * 4);
    unsigned short* qb  = (unsigned short*)alloc((size_t)MROWS * DIM * 2);
    unsigned short* kb  = (unsigned short*)alloc((size_t)MROWS * KVDIM * 2);
    unsigned short* vt  = (unsigned short*)alloc((size_t)MROWS * KVDIM * 2);
    unsigned short* ab  = (unsigned short*)alloc((size_t)MROWS * DIM * 2);

    const int T = 256;
    size_t n;

    n = (size_t)MROWS * DIM;
    cvt_bf16_kernel<<<dim3((unsigned)((n + T - 1) / T)), T, 0, stream>>>(x, xb, n);
    n = (size_t)DIM * DIM;
    cvtT_bf16_kernel<<<dim3((unsigned)((n + T - 1) / T)), T, 0, stream>>>(wq, wqT, DIM, DIM);
    n = (size_t)KVDIM * DIM;
    cvtT_bf16_kernel<<<dim3((unsigned)((n + T - 1) / T)), T, 0, stream>>>(wk, wkT, DIM, KVDIM);
    cvtT_bf16_kernel<<<dim3((unsigned)((n + T - 1) / T)), T, 0, stream>>>(wv, wvT, DIM, KVDIM);
    n = (size_t)DIM * DIM;
    cvtT_bf16_kernel<<<dim3((unsigned)((n + T - 1) / T)), T, 0, stream>>>(wo, woT, DIM, DIM);

    gemm_bf16_wmma<<<dim3(DIM / BN, MROWS / BM), 256, 0, stream>>>(xb, wqT, qf, MROWS, DIM, DIM);
    gemm_bf16_wmma<<<dim3(KVDIM / BN, MROWS / BM), 256, 0, stream>>>(xb, wkT, kf, MROWS, KVDIM, DIM);
    gemm_bf16_wmma<<<dim3(KVDIM / BN, MROWS / BM), 256, 0, stream>>>(xb, wvT, vf, MROWS, KVDIM, DIM);

    const float scale = 0.08838834764831845f;  // 1/sqrt(HEAD_DIM)
    n = (size_t)MROWS * NH * 64;
    rope_cvt_kernel<<<dim3((unsigned)((n + T - 1) / T)), T, 0, stream>>>(qf, qb, fc, fs, NH, scale, n);
    n = (size_t)MROWS * NKVH * 64;
    rope_cvt_kernel<<<dim3((unsigned)((n + T - 1) / T)), T, 0, stream>>>(kf, kb, fc, fs, NKVH, 1.0f, n);
    n = (size_t)MROWS * KVDIM;
    vtrans_kernel<<<dim3((unsigned)((n + T - 1) / T)), T, 0, stream>>>(vf, vt, n);

    attn_fwd_kernel<<<dim3(SEQ / BQ, NH, BSZ), 128, 0, stream>>>(qb, kb, vt, ab);

    gemm_bf16_wmma<<<dim3(DIM / BN, MROWS / BM), 256, 0, stream>>>(ab, woT, (float*)d_out, MROWS, DIM, DIM);
}
